// CausalMessagePassing_89558658056737
// MI455X (gfx1250) — matrix operand order repaired and checked by reading, not compiled, not run on hardware
//
#include <hip/hip_runtime.h>
#include <hip/hip_bf16.h>

#define BB 8
#define NN 2048
#define DD 256
#define HH 4
#define HDD 64

typedef __attribute__((ext_vector_type(16))) __bf16 v16bf;
typedef __attribute__((ext_vector_type(8)))  __bf16 v8bf;
typedef __attribute__((ext_vector_type(8)))  float  v8f;

// ---------------------------------------------------------------------------
// WMMA wrapper: D = A(16x32 bf16) * B(32x16 bf16) + C(16x16 f32)
// ---------------------------------------------------------------------------
__device__ __forceinline__ v8f wmma_bf16(v16bf a, v16bf b, v8f c) {
  return __builtin_amdgcn_wmma_f32_16x16x32_bf16(
      /*neg_a=*/false, a, /*neg_b=*/false, b,
      /*c_mod=*/(short)0, c, /*reuse_a=*/false, /*reuse_b=*/false);
}

// A-fragment (16-bit A 16x32): lane<16 -> K{0..7,16..23}, lane>=16 -> K{8..15,24..31}
__device__ __forceinline__ v16bf load16_split(const __bf16* p) {
  v8bf lo = *(const v8bf*)(p);
  v8bf hi = *(const v8bf*)(p + 16);
  v16bf a;
#pragma unroll
  for (int e = 0; e < 8; ++e) { a[e] = lo[e]; a[e + 8] = hi[e]; }
  return a;
}

__device__ __forceinline__ v16bf load_a_frag(const __bf16* base, int stride, int k0, int lane) {
  int m  = lane & 15;
  int kb = (lane >> 4) << 3;   // 0 or 8
  return load16_split(base + (size_t)m * stride + k0 + kb);
}

// B-fragment (16-bit B 32x16): lane<16 -> K{0..7,16..23}, lane>=16 -> K{8..15,24..31} (column n fixed).
__device__ __forceinline__ v16bf load16_contig(const __bf16* p) {
  v8bf lo = *(const v8bf*)(p);
  v8bf hi = *(const v8bf*)(p + 16);
  v16bf b;
#pragma unroll
  for (int e = 0; e < 8; ++e) { b[e] = lo[e]; b[e + 8] = hi[e]; }
  return b;
}

// B-fragment from a pre-transposed bf16 weight WT[col][K] (K contiguous).
__device__ __forceinline__ v16bf load_b_frag_wt(const __bf16* __restrict__ WT, int ldk,
                                                int k0, int n0, int lane) {
  int n  = lane & 15;
  int kb = (lane >> 4) << 3;   // 0 or 8
  return load16_contig(WT + (size_t)(n0 + n) * ldk + k0 + kb);
}

// Row reductions inside a 16-lane half-group (wave32; C-layout rows span 16 lanes)
__device__ __forceinline__ float rowmax16(float x) {
#pragma unroll
  for (int m = 1; m < 16; m <<= 1) x = fmaxf(x, __shfl_xor(x, m, 32));
  return x;
}
__device__ __forceinline__ float rowsum16(float x) {
#pragma unroll
  for (int m = 1; m < 16; m <<= 1) x += __shfl_xor(x, m, 32);
  return x;
}

// ---------------------------------------------------------------------------
// Kernel 0: one-time weight prep -> bf16, transposed to WT[col][K] layout.
// ---------------------------------------------------------------------------
__global__ __launch_bounds__(256) void prep_weights(
    const float* __restrict__ Wq, const float* __restrict__ Wk,
    const float* __restrict__ Wv, const float* __restrict__ Wo,
    const float* __restrict__ Wu,
    __bf16* __restrict__ qT, __bf16* __restrict__ kT, __bf16* __restrict__ vT,
    __bf16* __restrict__ oT, __bf16* __restrict__ uT) {
  int idx = blockIdx.x * 256 + threadIdx.x;
  if (idx < 4 * DD * DD) {
    int mat = idx >> 16;
    int loc = idx & 0xFFFF;
    int col = loc >> 8, kk = loc & 255;
    const float* src = (mat == 0) ? Wq : (mat == 1) ? Wk : (mat == 2) ? Wv : Wo;
    __bf16*      dst = (mat == 0) ? qT : (mat == 1) ? kT : (mat == 2) ? vT : oT;
    dst[(size_t)col * DD + kk] = (__bf16)src[(size_t)kk * DD + col];
  } else {
    int loc = idx - 4 * DD * DD;    // [0, 512*256)
    int col = loc >> 9, kk = loc & 511;
    uT[(size_t)col * (2 * DD) + kk] = (__bf16)Wu[(size_t)kk * DD + col];
  }
}

// ---------------------------------------------------------------------------
// Kernel 1: Q/K/V projections. Q,K -> bf16 [B,H,N,HD]; V -> bf16 [B,H,HD,N]
// Separate loops per matrix => scalar (SGPR) weight bases => global_load saddr.
// ---------------------------------------------------------------------------
__global__ __launch_bounds__(128) void qkv_kernel(
    const float* __restrict__ x,
    const __bf16* __restrict__ WqT, const float* __restrict__ bq,
    const __bf16* __restrict__ WkT, const float* __restrict__ bk,
    const __bf16* __restrict__ WvT, const float* __restrict__ bv,
    __bf16* __restrict__ qo, __bf16* __restrict__ ko, __bf16* __restrict__ vo) {
  __shared__ __align__(16) __bf16 xs[16][DD + 8];
  const int tid = threadIdx.x, lane = tid & 31, wid = tid >> 5;
  const int rowbase = blockIdx.x * 16;  // flat row over B*N

  for (int idx = tid; idx < 16 * DD; idx += 128) {
    int r = idx >> 8, c = idx & 255;
    xs[r][c] = (__bf16)x[(size_t)(rowbase + r) * DD + c];
  }
  __syncthreads();

  // Hoist all 8 A-fragments for this 16x256 tile (reused for Q, K and V)
  v16bf afr[8];
#pragma unroll
  for (int kc = 0; kc < 8; ++kc)
    afr[kc] = load_a_frag(&xs[0][0], DD + 8, kc * 32, lane);

  const int bi    = rowbase >> 11;          // batch
  const int nrow0 = rowbase & (NN - 1);     // row within batch
  const int rloc  = (lane >> 4) << 3;       // 0 or 8 (C-layout row offset)

  // ---- Q (pre-scaled by 1/sqrt(HD)) ----
  for (int t = wid; t < 16; t += 4) {
    const int n0 = t << 4;
    v8f acc = {};
#pragma unroll
    for (int kc = 0; kc < 8; ++kc)
      acc = wmma_bf16(afr[kc], load_b_frag_wt(WqT, DD, kc * 32, n0, lane), acc);
    const int col = n0 + (lane & 15);
    const float bias = bq[col];
    const int h = col >> 6, hd = col & 63;
    __bf16* dst = qo + ((((size_t)bi * HH + h) * NN + nrow0 + rloc) << 6) + hd;
#pragma unroll
    for (int vv = 0; vv < 8; ++vv)
      dst[(size_t)vv << 6] = (__bf16)((acc[vv] + bias) * 0.125f);
  }

  // ---- K ----
  for (int t = wid; t < 16; t += 4) {
    const int n0 = t << 4;
    v8f acc = {};
#pragma unroll
    for (int kc = 0; kc < 8; ++kc)
      acc = wmma_bf16(afr[kc], load_b_frag_wt(WkT, DD, kc * 32, n0, lane), acc);
    const int col = n0 + (lane & 15);
    const float bias = bk[col];
    const int h = col >> 6, hd = col & 63;
    __bf16* dst = ko + ((((size_t)bi * HH + h) * NN + nrow0 + rloc) << 6) + hd;
#pragma unroll
    for (int vv = 0; vv < 8; ++vv)
      dst[(size_t)vv << 6] = (__bf16)(acc[vv] + bias);
  }

  // ---- V (stored transposed [B,H,HD,N]) ----
  for (int t = wid; t < 16; t += 4) {
    const int n0 = t << 4;
    v8f acc = {};
#pragma unroll
    for (int kc = 0; kc < 8; ++kc)
      acc = wmma_bf16(afr[kc], load_b_frag_wt(WvT, DD, kc * 32, n0, lane), acc);
    const int col = n0 + (lane & 15);
    const float bias = bv[col];
    const int h = col >> 6, hd = col & 63;
    __bf16* dst = vo + (((size_t)bi * HH + h) * HDD + hd) * NN + nrow0 + rloc;
#pragma unroll
    for (int vv = 0; vv < 8; ++vv)
      dst[vv] = (__bf16)(acc[vv] + bias);
  }
}

// ---------------------------------------------------------------------------
// Kernel 2: causal flash attention. One wave per 16-query tile, 32 keys/iter.
// q,k: [B,H,N,HD]; v: [B,H,HD,N] (transposed). ctx out: bf16 [B,N,D].
// ---------------------------------------------------------------------------
__global__ __launch_bounds__(128) void attn_kernel(
    const __bf16* __restrict__ q, const __bf16* __restrict__ k,
    const __bf16* __restrict__ v, __bf16* __restrict__ ctx) {
  __shared__ __align__(16) __bf16 plds[4][16][40];  // per-wave P tile (16x32, padded)
  const int lane = threadIdx.x & 31, wid = threadIdx.x >> 5;
  const int unit = blockIdx.x * 4 + wid;          // [0, B*H*N/16)
  const int b  = unit >> 9;                       // / (H * N/16)
  const int h  = (unit >> 7) & (HH - 1);
  const int it = unit & 127;                      // query tile index
  const size_t headoff = (((size_t)b * HH + h) * NN) << 6;
  const __bf16* qp  = q + headoff;
  const __bf16* kp  = k + headoff;
  const __bf16* vtp = v + (((size_t)b * HH + h) * HDD) * NN;  // [HD][N]

  const int m     = lane & 15;
  const int kb8   = (lane >> 4) << 3;
  const int kb16  = (lane >> 4) << 3;
  const int qrow0 = it << 4;

  // Hoisted Q A-fragments (HD=64 -> 2 chunks of K=32)
  v16bf qa[2];
#pragma unroll
  for (int c = 0; c < 2; ++c)
    qa[c] = load16_split(qp + (size_t)(qrow0 + m) * HDD + c * 32 + kb8);

  float mrow[8], lrow[8];
  v8f o[4] = {v8f{}, v8f{}, v8f{}, v8f{}};
#pragma unroll
  for (int i = 0; i < 8; ++i) { mrow[i] = -3.0e38f; lrow[i] = 0.0f; }

  const int nb = ((qrow0 + 15) >> 5) + 1;  // key blocks up to the diagonal
  for (int jb = 0; jb < nb; ++jb) {
    const int j0 = jb << 5;
    v8f s[2];
#pragma unroll
    for (int half = 0; half < 2; ++half) {
      v8f acc = {};
      const int key = j0 + (half << 4) + m;  // B-frag column = key index
#pragma unroll
      for (int c = 0; c < 2; ++c) {
        v16bf bf = load16_contig(kp + (size_t)key * HDD + c * 32 + kb16);
        acc = wmma_bf16(qa[c], bf, acc);
      }
      if (j0 + (half << 4) + 15 > qrow0) {  // diagonal block: causal mask
#pragma unroll
        for (int vv = 0; vv < 8; ++vv) {
          int rowg = qrow0 + vv + kb8;
          int colg = j0 + (half << 4) + m;
          if (colg > rowg) acc[vv] = -3.0e38f;
        }
      }
      s[half] = acc;
    }

    // Online softmax across the 32-key block
#pragma unroll
    for (int vv = 0; vv < 8; ++vv) {
      float rm    = rowmax16(fmaxf(s[0][vv], s[1][vv]));
      float mnew  = fmaxf(mrow[vv], rm);
      float alpha = __expf(mrow[vv] - mnew);
      float p0    = __expf(s[0][vv] - mnew);
      float p1    = __expf(s[1][vv] - mnew);
      lrow[vv] = lrow[vv] * alpha + rowsum16(p0 + p1);
      mrow[vv] = mnew;
#pragma unroll
      for (int t = 0; t < 4; ++t) o[t][vv] *= alpha;
      int rloc = vv + kb8;
      plds[wid][rloc][m]      = (__bf16)p0;   // C-layout -> A-layout via LDS
      plds[wid][rloc][16 + m] = (__bf16)p1;
    }

    // O += P(16x32) @ V(32x64); V is transposed so B-frags are contiguous
    v16bf pa = load16_split(&plds[wid][m][kb8]);
#pragma unroll
    for (int t = 0; t < 4; ++t) {
      v16bf bf = load16_contig(vtp + (size_t)(t * 16 + m) * NN + j0 + kb16);
      o[t] = wmma_bf16(pa, bf, o[t]);
    }
  }

  // Epilogue: normalize, write ctx as [B, N, D] bf16
#pragma unroll
  for (int vv = 0; vv < 8; ++vv) {
    float inv = 1.0f / lrow[vv];
    int grow = b * NN + qrow0 + vv + kb8;
#pragma unroll
    for (int t = 0; t < 4; ++t) {
      int col = (h << 6) + (t << 4) + m;
      ctx[(size_t)grow * DD + col] = (__bf16)(o[t][vv] * inv);
    }
  }
}

// ---------------------------------------------------------------------------
// Kernel 3: messages = ctx @ Wo + bo -> bf16 [B, N, D]
// ---------------------------------------------------------------------------
__global__ __launch_bounds__(128) void msg_kernel(
    const __bf16* __restrict__ ctx, const __bf16* __restrict__ WoT,
    const float* __restrict__ bo, __bf16* __restrict__ msg) {
  __shared__ __align__(16) __bf16 cs[16][DD + 8];
  const int tid = threadIdx.x, lane = tid & 31, wid = tid >> 5;
  const int rowbase = blockIdx.x * 16;
  for (int idx = tid; idx < 16 * DD; idx += 128) {
    int r = idx >> 8, c = idx & 255;
    cs[r][c] = ctx[(size_t)(rowbase + r) * DD + c];
  }
  __syncthreads();

  v16bf afr[8];
#pragma unroll
  for (int kc = 0; kc < 8; ++kc)
    afr[kc] = load_a_frag(&cs[0][0], DD + 8, kc * 32, lane);

  for (int t = wid; t < 16; t += 4) {
    const int n0 = t << 4;
    v8f acc = {};
#pragma unroll
    for (int kc = 0; kc < 8; ++kc)
      acc = wmma_bf16(afr[kc], load_b_frag_wt(WoT, DD, kc * 32, n0, lane), acc);
    const int col = n0 + (lane & 15);
    const float bias = bo[col];
#pragma unroll
    for (int vv = 0; vv < 8; ++vv) {
      int grow = rowbase + vv + ((lane >> 4) << 3);
      msg[(size_t)grow * DD + col] = (__bf16)(acc[vv] + bias);
    }
  }
}

// ---------------------------------------------------------------------------
// Kernel 4: out = relu([x, msg] @ Wu + bu) -> f32 [B, N, D]  (K = 512 GEMM)
// ---------------------------------------------------------------------------
__global__ __launch_bounds__(128) void ffn_kernel(
    const float* __restrict__ x, const __bf16* __restrict__ msg,
    const __bf16* __restrict__ WuT, const float* __restrict__ bu,
    float* __restrict__ out) {
  __shared__ __align__(16) __bf16 as[16][2 * DD + 8];
  const int tid = threadIdx.x, lane = tid & 31, wid = tid >> 5;
  const int rowbase = blockIdx.x * 16;
  for (int idx = tid; idx < 16 * DD; idx += 128) {
    int r = idx >> 8, c = idx & 255;
    as[r][c]      = (__bf16)x[(size_t)(rowbase + r) * DD + c];
    as[r][DD + c] = msg[(size_t)(rowbase + r) * DD + c];
  }
  __syncthreads();
  for (int t = wid; t < 16; t += 4) {
    const int n0 = t << 4;
    v8f acc = {};
#pragma unroll
    for (int kc = 0; kc < 16; ++kc) {
      v16bf a = load_a_frag(&as[0][0], 2 * DD + 8, kc * 32, lane);
      v16bf b = load_b_frag_wt(WuT, 2 * DD, kc * 32, n0, lane);
      acc = wmma_bf16(a, b, acc);
    }
    const int col = n0 + (lane & 15);
    const float bias = bu[col];
#pragma unroll
    for (int vv = 0; vv < 8; ++vv) {
      int grow = rowbase + vv + ((lane >> 4) << 3);
      out[(size_t)grow * DD + col] = fmaxf(acc[vv] + bias, 0.0f);
    }
  }
}

// ---------------------------------------------------------------------------
extern "C" void kernel_launch(void* const* d_in, const int* in_sizes, int n_in,
                              void* d_out, int out_size, void* d_ws, size_t ws_size,
                              hipStream_t stream) {
  const float* x  = (const float*)d_in[0];
  // d_in[1] = causal_mask: implemented analytically, not read
  const float* Wq = (const float*)d_in[2];
  const float* bq = (const float*)d_in[3];
  const float* Wk = (const float*)d_in[4];
  const float* bk = (const float*)d_in[5];
  const float* Wv = (const float*)d_in[6];
  const float* bv = (const float*)d_in[7];
  const float* Wo = (const float*)d_in[8];
  const float* bo = (const float*)d_in[9];
  const float* Wu = (const float*)d_in[10];
  const float* bu = (const float*)d_in[11];
  float* out = (float*)d_out;

  const size_t perMat = (size_t)BB * HH * NN * HDD;  // 4 Mi elements per bf16 buffer
  __bf16* qws = (__bf16*)d_ws;
  __bf16* kws = qws + perMat;
  __bf16* vws = kws + perMat;      // stored transposed [B,H,HD,N]
  __bf16* ctx = vws + perMat;
  __bf16* msg = ctx + perMat;
  __bf16* wqT = msg + perMat;
  __bf16* wkT = wqT + (size_t)DD * DD;
  __bf16* wvT = wkT + (size_t)DD * DD;
  __bf16* woT = wvT + (size_t)DD * DD;
  __bf16* wuT = woT + (size_t)DD * DD;   // 512*256

  prep_weights<<<1536, 256, 0, stream>>>(Wq, Wk, Wv, Wo, Wu, wqT, wkT, wvT, woT, wuT);

  const int rowTiles = BB * NN / 16;  // 1024
  qkv_kernel<<<rowTiles, 128, 0, stream>>>(x, wqT, bq, wkT, bk, wvT, bv, qws, kws, vws);
  attn_kernel<<<BB * HH * (NN / 16) / 4, 128, 0, stream>>>(qws, kws, vws, ctx);
  msg_kernel<<<rowTiles, 128, 0, stream>>>(ctx, woT, bo, msg);
  ffn_kernel<<<rowTiles, 128, 0, stream>>>(x, msg, wuT, bu, out);
}
